// MultiHeadAttention_42356967473813
// MI455X (gfx1250) — matrix operand order repaired and checked by reading, not compiled, not run on hardware
//
#include <hip/hip_runtime.h>
#include <hip/hip_bf16.h>

// Problem constants (from reference)
constexpr int Bb = 4;
constexpr int Ss = 2048;
constexpr int Ee = 1024;
constexpr int Hh = 16;
constexpr int Dd = 64;
constexpr int E3 = 3 * Ee; // 3072

typedef __attribute__((ext_vector_type(16))) __bf16 v16bf;
typedef __attribute__((ext_vector_type(8)))  __bf16 v8bf;
typedef __attribute__((ext_vector_type(8)))  float  v8f;

__device__ __forceinline__ v8f vzero() {
  v8f z = {0.f, 0.f, 0.f, 0.f, 0.f, 0.f, 0.f, 0.f};
  return z;
}

__device__ __forceinline__ v8f wmma_bf16(v16bf a, v16bf b, v8f c) {
  // D = A(16x32 bf16) x B(32x16 bf16) + C(16x16 f32)
  return __builtin_amdgcn_wmma_f32_16x16x32_bf16(
      /*neg_a=*/false, a, /*neg_b=*/false, b,
      /*c_mod=*/(short)0, c, /*reuse_a=*/false, /*reuse_b=*/false);
}

// A-matrix fragment (16x32, row-major source with `stride` elements per row).
// ISA layout: lanes 0-15 hold row M=lane, K={0..7} then {16..23};
//             lanes 16-31 hold row M=lane-16, K={8..15} then {24..31}.
__device__ __forceinline__ v16bf load_frag_a(const __bf16* __restrict__ p,
                                             int stride, int k0, int lane) {
  const int lh = (lane >> 4) & 1;
  const int r  = lane & 15;
  const __bf16* base = p + (size_t)r * stride + k0 + lh * 8;
  v8bf lo = *(const v8bf*)(base);        // K = k0 + lh*8 .. +7
  v8bf hi = *(const v8bf*)(base + 16);   // K = k0 + 16 + lh*8 .. +7
  return __builtin_shufflevector(lo, hi, 0,1,2,3,4,5,6,7,8,9,10,11,12,13,14,15);
}

// B-matrix fragment (32x16) loaded from an N-major (transposed) buffer:
// row n of the buffer is column n of B, contiguous over K.
// ISA layout: lanes 0-15 hold col N=lane, K={0..7} then {16..23};
//             lanes 16-31 hold K={8..15} then {24..31}.
__device__ __forceinline__ v16bf load_frag_b(const __bf16* __restrict__ p,
                                             int stride, int k0, int lane) {
  const int lh = (lane >> 4) & 1;
  const int n  = lane & 15;
  const __bf16* base = p + (size_t)n * stride + k0 + lh * 8;
  v8bf lo = *(const v8bf*)(base);        // K = k0 + lh*8 .. +7
  v8bf hi = *(const v8bf*)(base + 16);   // K = k0 + 16 + lh*8 .. +7
  return __builtin_shufflevector(lo, hi, 0,1,2,3,4,5,6,7,8,9,10,11,12,13,14,15);
}

// ---------------- conversion kernels ----------------

__global__ void k_cvt(const float* __restrict__ s, __bf16* __restrict__ d, int n) {
  for (int i = blockIdx.x * blockDim.x + threadIdx.x; i < n;
       i += gridDim.x * blockDim.x)
    d[i] = (__bf16)s[i];
}

// src [rows, cols] f32 -> dst [cols, rows] bf16 (transpose + convert)
__global__ void k_tr_cvt(const float* __restrict__ s, __bf16* __restrict__ d,
                         int rows, int cols) {
  int n = rows * cols;
  for (int i = blockIdx.x * blockDim.x + threadIdx.x; i < n;
       i += gridDim.x * blockDim.x) {
    int r = i / cols, c = i - r * cols;
    d[(size_t)c * rows + r] = (__bf16)s[i];
  }
}

// ---------------- QKV GEMM with fused scatter epilogue ----------------
// X [M=8192, K=1024] bf16; Wt [N=3072, K=1024] bf16 (transposed); bias f32[3072]
// Writes Q[b,h,s,d]*0.125, K[b,h,t,d], Vt[b,h,d,t] in bf16.
__global__ __launch_bounds__(256)
void k_gemm_qkv(const __bf16* __restrict__ X, const __bf16* __restrict__ Wt,
                const float* __restrict__ bias,
                __bf16* __restrict__ Q, __bf16* __restrict__ Kc,
                __bf16* __restrict__ Vt) {
  const int lane = threadIdx.x & 31;
  const int wid  = blockIdx.x * 8 + (threadIdx.x >> 5);
  const int nTiles = E3 / 64;              // 48
  const int mi = wid / nTiles, ni = wid - mi * nTiles;
  const int m0 = mi * 32, n0 = ni * 64;

  v8f acc[2][4];
#pragma unroll
  for (int i = 0; i < 2; ++i)
#pragma unroll
    for (int j = 0; j < 4; ++j) acc[i][j] = vzero();

  for (int k0 = 0; k0 < Ee; k0 += 32) {
    v16bf a0 = load_frag_a(X + (size_t)m0 * Ee, Ee, k0, lane);
    v16bf a1 = load_frag_a(X + (size_t)(m0 + 16) * Ee, Ee, k0, lane);
#pragma unroll
    for (int j = 0; j < 4; ++j) {
      v16bf bw = load_frag_b(Wt + (size_t)(n0 + j * 16) * Ee, Ee, k0, lane);
      acc[0][j] = wmma_bf16(a0, bw, acc[0][j]);
      acc[1][j] = wmma_bf16(a1, bw, acc[1][j]);
    }
  }

  const int lh = lane >> 4, nn = lane & 15;
#pragma unroll
  for (int i = 0; i < 2; ++i)
#pragma unroll
    for (int j = 0; j < 4; ++j) {
      const int col = n0 + j * 16 + nn;
      const int h = col / (3 * Dd);
      const int rr = col - h * (3 * Dd);
      const int which = rr >> 6;           // 0=q 1=k 2=v
      const int d = rr & (Dd - 1);
      const float bv = bias[col];
#pragma unroll
      for (int v = 0; v < 8; ++v) {
        const int row = m0 + i * 16 + v + lh * 8;
        const int b_ = row >> 11;          // row / S
        const int s_ = row & (Ss - 1);
        const float val = acc[i][j][v] + bv;
        const size_t bh = (size_t)(b_ * Hh + h);
        if (which == 0)
          Q[(bh * Ss + s_) * Dd + d] = (__bf16)(val * 0.125f);  // 1/sqrt(64)
        else if (which == 1)
          Kc[(bh * Ss + s_) * Dd + d] = (__bf16)val;
        else
          Vt[(bh * Dd + d) * Ss + s_] = (__bf16)val;
      }
    }
}

// ---------------- flash attention (one wave = 16 query rows) ----------------
__global__ __launch_bounds__(256)
void k_attn(const __bf16* __restrict__ Q, const __bf16* __restrict__ Kc,
            const __bf16* __restrict__ Vt, __bf16* __restrict__ Out) {
  __shared__ __bf16 pbuf[8 * 16 * 64];     // 2KB per wave staging for P tiles
  const int lane = threadIdx.x & 31;
  const int wv   = threadIdx.x >> 5;
  const int wid  = blockIdx.x * 8 + wv;
  const int mTiles = Ss / 16;              // 128
  const int bh = wid / mTiles;
  const int m0 = (wid - bh * mTiles) * 16;
  const int h = bh % Hh, b = bh / Hh;

  const __bf16* Qb = Q  + ((size_t)bh * Ss + m0) * Dd;
  const __bf16* Kb = Kc + (size_t)bh * Ss * Dd;
  const __bf16* Vb = Vt + (size_t)bh * Dd * Ss;
  __bf16* ps = &pbuf[wv * 16 * 64];

  // Q fragments for this row tile (16 x 64 = two 16x32 A-frags); scale baked in.
  v16bf aq0 = load_frag_a(Qb, Dd, 0, lane);
  v16bf aq1 = load_frag_a(Qb, Dd, 32, lane);

  v8f acc[4];
#pragma unroll
  for (int j = 0; j < 4; ++j) acc[j] = vzero();
  float mrun[8], srun[8];
#pragma unroll
  for (int v = 0; v < 8; ++v) { mrun[v] = -1e30f; srun[v] = 0.f; }

  const int lh = lane >> 4, nn = lane & 15;

  for (int t0 = 0; t0 < Ss; t0 += 64) {
    // ---- scores: S[16 x 64] = Q(16x64) x K^T(64x64chunk), k-dim = D
    v8f st[4];
#pragma unroll
    for (int j = 0; j < 4; ++j) {
      const __bf16* kt = Kb + (size_t)(t0 + j * 16) * Dd;
      v16bf bk0 = load_frag_b(kt, Dd, 0, lane);
      v16bf bk1 = load_frag_b(kt, Dd, 32, lane);
      v8f z = vzero();
      z = wmma_bf16(aq0, bk0, z);
      z = wmma_bf16(aq1, bk1, z);
      st[j] = z;
    }
    // ---- online softmax update (row r = v + lh*8 lives in one VGPR slot
    //      across the 16 lanes of this half -> butterfly over masks 1,2,4,8)
    v8f mt = st[0];
#pragma unroll
    for (int j = 1; j < 4; ++j)
#pragma unroll
      for (int v = 0; v < 8; ++v) mt[v] = fmaxf(mt[v], st[j][v]);
#pragma unroll
    for (int mask = 1; mask <= 8; mask <<= 1)
#pragma unroll
      for (int v = 0; v < 8; ++v)
        mt[v] = fmaxf(mt[v], __shfl_xor(mt[v], mask, 32));

    float fs[8];
#pragma unroll
    for (int v = 0; v < 8; ++v) {
      const float mnew = fmaxf(mrun[v], mt[v]);
      fs[v] = __expf(mrun[v] - mnew);
      mrun[v] = mnew;
    }
#pragma unroll
    for (int j = 0; j < 4; ++j)
#pragma unroll
      for (int v = 0; v < 8; ++v) st[j][v] = __expf(st[j][v] - mrun[v]);

    v8f su;
#pragma unroll
    for (int v = 0; v < 8; ++v)
      su[v] = st[0][v] + st[1][v] + st[2][v] + st[3][v];
#pragma unroll
    for (int mask = 1; mask <= 8; mask <<= 1)
#pragma unroll
      for (int v = 0; v < 8; ++v) su[v] += __shfl_xor(su[v], mask, 32);
#pragma unroll
    for (int v = 0; v < 8; ++v) srun[v] = srun[v] * fs[v] + su[v];
#pragma unroll
    for (int j = 0; j < 4; ++j)
#pragma unroll
      for (int v = 0; v < 8; ++v) acc[j][v] *= fs[v];

    // ---- re-layout P (C-layout) -> A-fragments via per-wave LDS tile
    //      (same-wave RAW: DS ops are in-order; wait DScnt before reload)
#pragma unroll
    for (int j = 0; j < 4; ++j)
#pragma unroll
      for (int v = 0; v < 8; ++v)
        ps[(v + lh * 8) * 64 + j * 16 + nn] = (__bf16)st[j][v];
    asm volatile("s_wait_dscnt 0" ::: "memory");
    v16bf ap0 = load_frag_a(ps, 64, 0, lane);
    v16bf ap1 = load_frag_a(ps, 64, 32, lane);

    // ---- O += P(16x64) x V(64chunk x 64), B-cols contiguous in Vt[d][t]
#pragma unroll
    for (int j = 0; j < 4; ++j) {
      const __bf16* vt = Vb + (size_t)(j * 16) * Ss + t0;
      v16bf bv0 = load_frag_b(vt, Ss, 0, lane);
      v16bf bv1 = load_frag_b(vt, Ss, 32, lane);
      acc[j] = wmma_bf16(ap0, bv0, acc[j]);
      acc[j] = wmma_bf16(ap1, bv1, acc[j]);
    }
  }

  // ---- epilogue: Out[b, s, h*64 + d] = acc / rowsum  (bf16)
#pragma unroll
  for (int j = 0; j < 4; ++j)
#pragma unroll
    for (int v = 0; v < 8; ++v) {
      const int row = m0 + v + lh * 8;
      const float o = acc[j][v] / srun[v];
      Out[((size_t)b * Ss + row) * Ee + h * Dd + j * 16 + nn] = (__bf16)o;
    }
}

// ---------------- output GEMM: f32 out + bias ----------------
__global__ __launch_bounds__(256)
void k_gemm_out(const __bf16* __restrict__ A, const __bf16* __restrict__ Wt,
                const float* __restrict__ bias, float* __restrict__ Out) {
  const int lane = threadIdx.x & 31;
  const int wid  = blockIdx.x * 8 + (threadIdx.x >> 5);
  const int nTiles = Ee / 64;              // 16
  const int mi = wid / nTiles, ni = wid - mi * nTiles;
  const int m0 = mi * 32, n0 = ni * 64;

  v8f acc[2][4];
#pragma unroll
  for (int i = 0; i < 2; ++i)
#pragma unroll
    for (int j = 0; j < 4; ++j) acc[i][j] = vzero();

  for (int k0 = 0; k0 < Ee; k0 += 32) {
    v16bf a0 = load_frag_a(A + (size_t)m0 * Ee, Ee, k0, lane);
    v16bf a1 = load_frag_a(A + (size_t)(m0 + 16) * Ee, Ee, k0, lane);
#pragma unroll
    for (int j = 0; j < 4; ++j) {
      v16bf bw = load_frag_b(Wt + (size_t)(n0 + j * 16) * Ee, Ee, k0, lane);
      acc[0][j] = wmma_bf16(a0, bw, acc[0][j]);
      acc[1][j] = wmma_bf16(a1, bw, acc[1][j]);
    }
  }

  const int lh = lane >> 4, nn = lane & 15;
#pragma unroll
  for (int i = 0; i < 2; ++i)
#pragma unroll
    for (int j = 0; j < 4; ++j) {
      const int col = n0 + j * 16 + nn;
      const float bv = bias[col];
#pragma unroll
      for (int v = 0; v < 8; ++v) {
        const int row = m0 + i * 16 + v + lh * 8;
        Out[(size_t)row * Ee + col] = acc[i][j][v] + bv;
      }
    }
}

// ---------------- launcher ----------------
extern "C" void kernel_launch(void* const* d_in, const int* in_sizes, int n_in,
                              void* d_out, int out_size, void* d_ws, size_t ws_size,
                              hipStream_t stream) {
  (void)in_sizes; (void)n_in; (void)out_size; (void)ws_size;
  const float* x     = (const float*)d_in[0];
  const float* w_qkv = (const float*)d_in[1];
  const float* b_qkv = (const float*)d_in[2];
  const float* w_out = (const float*)d_in[3];
  const float* b_out = (const float*)d_in[4];
  float* out = (float*)d_out;

  char* ws = (char*)d_ws;
  size_t off = 0;
  auto take = [&](size_t bytes) -> void* {
    void* p = ws + off;
    off += (bytes + 255) & ~(size_t)255;
    return p;
  };
  const size_t M = (size_t)Bb * Ss;                // 8192
  __bf16* xb     = (__bf16*)take(M * Ee * 2);      // x in bf16
  __bf16* wqkvT  = (__bf16*)take((size_t)E3 * Ee * 2);
  __bf16* woutT  = (__bf16*)take((size_t)Ee * Ee * 2);
  __bf16* Qw     = (__bf16*)take((size_t)Bb * Hh * Ss * Dd * 2);
  __bf16* Kw     = (__bf16*)take((size_t)Bb * Hh * Ss * Dd * 2);
  __bf16* Vw     = (__bf16*)take((size_t)Bb * Hh * Dd * Ss * 2);
  __bf16* attn   = (__bf16*)take(M * Ee * 2);

  k_cvt<<<2048, 256, 0, stream>>>(x, xb, (int)(M * Ee));
  k_tr_cvt<<<2048, 256, 0, stream>>>(w_qkv, wqkvT, Ee, E3);
  k_tr_cvt<<<1024, 256, 0, stream>>>(w_out, woutT, Ee, Ee);

  // (M/32)*(3072/64) = 256*48 = 12288 waves / 8 per block
  k_gemm_qkv<<<1536, 256, 0, stream>>>(xb, wqkvT, b_qkv, Qw, Kw, Vw);

  // B*H*(S/16) = 8192 waves / 8 per block
  k_attn<<<1024, 256, 0, stream>>>(Qw, Kw, Vw, attn);

  // (M/32)*(1024/64) = 256*16 = 4096 waves / 8 per block
  k_gemm_out<<<512, 256, 0, stream>>>(attn, woutT, b_out, out);
}
